// CrossRPEAttention_14173392076854
// MI455X (gfx1250) — matrix operand order repaired and checked by reading, not compiled, not run on hardware
//
#include <hip/hip_runtime.h>
#include <hip/hip_bf16.h>

// ---------------- problem constants ----------------
#define BB    16
#define HH    16
#define NPP   512      // NP (tokens after truncation)
#define HD    64
#define DIMV  1024
#define NBK   7        // rpe buckets
#define EE    23       // ceil(sqrt(512))

typedef __attribute__((ext_vector_type(16))) __bf16 v16bf;
typedef __attribute__((ext_vector_type(8)))  __bf16 v8bf;
typedef __attribute__((ext_vector_type(8)))  float  v8f;

// Assemble a 16-half WMMA fragment from two contiguous 16-byte chunks.
static __device__ __forceinline__ v16bf frag16(const __bf16* p0, const __bf16* p1) {
  v8bf a = *(const v8bf*)p0;
  v8bf b = *(const v8bf*)p1;
  return __builtin_shufflevector(a, b, 0,1,2,3,4,5,6,7,8,9,10,11,12,13,14,15);
}

static __device__ __forceinline__ v8f wmma_bf16(v16bf a, v16bf b, v8f c) {
  return __builtin_amdgcn_wmma_f32_16x16x32_bf16(false, a, false, b, (short)0, c, false, false);
}

// ---------------- prep kernels ----------------
__global__ void k_cvt_bf16(__bf16* __restrict__ dst, const float* __restrict__ src, int n) {
  int i = blockIdx.x * 256 + threadIdx.x;
  if (i < n) dst[i] = (__bf16)src[i];
}

// x: (16,1024,1024) f32 -> xb: (16,512,1024) bf16 (truncate tokens to NP=512)
__global__ void k_cvt_x(__bf16* __restrict__ xb, const float* __restrict__ x) {
  int i = blockIdx.x * 256 + threadIdx.x;          // 0 .. 16*512*1024-1
  int c = i & (DIMV - 1);
  int p = (i >> 10) & (NPP - 1);
  int b = i >> 19;
  xb[i] = (__bf16)x[((size_t)b * 1024 + p) * 1024 + c];
}

// Static iRPE 'euc' bucket table, double math to match numpy. idx.min()==0.
__global__ void k_bucket(unsigned char* __restrict__ bkt) {
  int idx = blockIdx.x * 256 + threadIdx.x;        // 0 .. 512*512-1
  int i = idx >> 9, j = idx & (NPP - 1);
  int pi0 = i / EE, pi1 = i % EE;
  int pj0 = j / EE, pj1 = j % EE;
  double d0 = (double)(pi0 - pj0), d1 = (double)(pi1 - pj1);
  double dis = rint(sqrt(d0 * d0 + d1 * d1));
  double val;
  if (dis <= 1.9) {
    val = rint(dis);
  } else {
    double far = rint(1.9 + log(fmax(dis, 1e-9) / 1.9) / log(15.2 / 1.9) * (3.8 - 1.9));
    val = fmin(far, 3.8);
  }
  bkt[idx] = (unsigned char)(int)val;              // truncation, as in .astype(int32)
}

// ---------------- fused QKV projection (bf16 WMMA) ----------------
// grid (256, 4), 256 threads = 8 waves; wave -> 32x32 tile of each of q,k,v.
// Per K-step: 2 A-frags + 6 B-frags (16 x b128) feed 12 WMMAs (1.33 loads/WMMA).
__global__ void __launch_bounds__(256)
k_qkv(const __bf16* __restrict__ xb,
      const __bf16* __restrict__ wq, const __bf16* __restrict__ wk,
      const __bf16* __restrict__ wv,
      __bf16* __restrict__ qh, __bf16* __restrict__ kh, __bf16* __restrict__ vT) {
  const int lane = threadIdx.x & 31;
  const int wave = threadIdx.x >> 5;
  const int mr   = lane & 15;
  const int half = lane >> 4;
  const int m_base = blockIdx.x * 32;
  const int j0     = (blockIdx.y * 8 + wave) * 32;

  const __bf16* ar[2];
  ar[0] = xb + (size_t)(m_base + mr) * DIMV;
  ar[1] = ar[0] + 16 * DIMV;
  const __bf16* br[3][2];
  br[0][0] = wq + (size_t)(j0 + mr) * DIMV;  br[0][1] = br[0][0] + 16 * DIMV;
  br[1][0] = wk + (size_t)(j0 + mr) * DIMV;  br[1][1] = br[1][0] + 16 * DIMV;
  br[2][0] = wv + (size_t)(j0 + mr) * DIMV;  br[2][1] = br[2][0] + 16 * DIMV;

  v8f acc[3][2][2];   // [weight][mi][ni]
  #pragma unroll
  for (int w = 0; w < 3; ++w)
    #pragma unroll
    for (int mi = 0; mi < 2; ++mi)
      #pragma unroll
      for (int ni = 0; ni < 2; ++ni) acc[w][mi][ni] = (v8f){};

  for (int kk = 0; kk < DIMV; kk += 32) {
    v16bf a[2], f[3][2];
    #pragma unroll
    for (int mi = 0; mi < 2; ++mi)
      a[mi] = frag16(ar[mi] + kk + half * 8, ar[mi] + kk + 16 + half * 8);
    #pragma unroll
    for (int w = 0; w < 3; ++w)
      #pragma unroll
      for (int ni = 0; ni < 2; ++ni)
        f[w][ni] = frag16(br[w][ni] + kk + half * 8, br[w][ni] + kk + 16 + half * 8);
    #pragma unroll
    for (int w = 0; w < 3; ++w)
      #pragma unroll
      for (int mi = 0; mi < 2; ++mi)
        #pragma unroll
        for (int ni = 0; ni < 2; ++ni)
          acc[w][mi][ni] = wmma_bf16(a[mi], f[w][ni], acc[w][mi][ni]);
  }

  const int h  = j0 >> 6;                     // 32-col tile lies in one head (j0%64 in {0,32})
  const int d0 = j0 & 63;
  #pragma unroll
  for (int mi = 0; mi < 2; ++mi) {
    const int Mb = m_base + mi * 16 + 8 * half;    // global row for r=0
    const int bb = Mb >> 9;
    const int p0 = Mb & (NPP - 1);
    #pragma unroll
    for (int ni = 0; ni < 2; ++ni) {
      const int d = d0 + ni * 16 + mr;
      __bf16* qp = qh + ((size_t)(bb * HH + h) * NPP + p0) * HD + d;
      __bf16* kp = kh + ((size_t)(bb * HH + h) * NPP + p0) * HD + d;
      #pragma unroll
      for (int r = 0; r < 8; ++r) {
        qp[r * HD] = (__bf16)acc[0][mi][ni][r];
        kp[r * HD] = (__bf16)acc[1][mi][ni][r];
      }
      // v stored transposed (d, p): 8 contiguous p values -> one b128 store
      __bf16* vp = vT + ((size_t)(bb * HH + h) * HD + d) * NPP + p0;
      v8bf vv;
      #pragma unroll
      for (int r = 0; r < 8; ++r) vv[r] = (__bf16)acc[2][mi][ni][r];
      *(v8bf*)vp = vv;
    }
  }
}

// ---------------- fused attention (scores + RPE bias + softmax + PV) ----------------
// grid 2048 = B*H*8 row-blocks of 64 rows; 128 threads = 4 waves, 16 rows/wave.
// Dynamic LDS (194 KB): 4x 16x512 f32 score strips, 4x 16x512 bf16 prob strips, lt.
__global__ void __launch_bounds__(128)
k_attn(const __bf16* __restrict__ qh, const __bf16* __restrict__ kh,
       const __bf16* __restrict__ vT, const float* __restrict__ rpe_w,
       const unsigned char* __restrict__ bkt, __bf16* __restrict__ ao) {
  extern __shared__ float smem[];
  const int lane = threadIdx.x & 31;
  const int wave = threadIdx.x >> 5;
  const int bh = blockIdx.x >> 3;                   // 0..255
  const int rb = blockIdx.x & 7;                    // 64-row block
  const int b = bh >> 4, h = bh & 15;
  const int p0 = rb * 64 + wave * 16;               // this wave's strip start row

  float*  S  = smem + wave * 16 * NPP;                              // 16x512 f32 scores
  __bf16* P  = (__bf16*)(smem + 4 * 16 * NPP) + wave * 16 * NPP;    // 16x512 bf16 probs
  float*  LT = smem + 4 * 16 * NPP + 2 * 16 * NPP + wave * 16 * 8;  // 16x8 lt values

  const __bf16* qbase = qh + ((size_t)bh * NPP + p0) * HD;
  const __bf16* kbase = kh + (size_t)bh * NPP * HD;
  const __bf16* vbase = vT + (size_t)bh * HD * NPP;

  // lt[m,t] = q[m,:] . rpe_w[:,t]   (lanes 0..15, one row each)
  if (lane < 16) {
    const __bf16* qr = qbase + lane * HD;
    float acc[NBK];
    #pragma unroll
    for (int t = 0; t < NBK; ++t) acc[t] = 0.f;
    for (int d = 0; d < HD; ++d) {
      float qv = (float)qr[d];
      #pragma unroll
      for (int t = 0; t < NBK; ++t) acc[t] += qv * rpe_w[d * NBK + t];
    }
    #pragma unroll
    for (int t = 0; t < NBK; ++t) LT[lane * 8 + t] = acc[t];
    LT[lane * 8 + 7] = 0.f;
  }
  __syncthreads();

  const int mr = lane & 15, half = lane >> 4;

  // Q strip A-fragments for both K-steps (HD=64 -> 2 x K32)
  const __bf16* arow = qbase + mr * HD;
  v16bf a0 = frag16(arow + half * 8,      arow + 16 + half * 8);
  v16bf a1 = frag16(arow + 32 + half * 8, arow + 48 + half * 8);

  // S = (q k^T) * scale + bias, staged in LDS
  for (int jt = 0; jt < 32; ++jt) {
    const int j0 = jt * 16;
    const __bf16* krow = kbase + (size_t)(j0 + mr) * HD;
    v16bf b0 = frag16(krow + half * 8,      krow + 16 + half * 8);
    v16bf b1 = frag16(krow + 32 + half * 8, krow + 48 + half * 8);
    v8f c = (v8f){};
    c = wmma_bf16(a0, b0, c);
    c = wmma_bf16(a1, b1, c);
    #pragma unroll
    for (int r = 0; r < 8; ++r) {
      int m  = r + 8 * half;
      int bu = bkt[(size_t)(p0 + m) * NPP + j0 + mr];
      S[m * NPP + j0 + mr] = c[r] * 0.125f + LT[m * 8 + bu];
    }
  }
  __syncthreads();

  // per-row softmax into the separate bf16 prob strip
  for (int m = 0; m < 16; ++m) {
    float*  row  = S + m * NPP;
    __bf16* prow = P + m * NPP;
    float mx = -3.0e38f;
    for (int c4 = 0; c4 < 16; ++c4) mx = fmaxf(mx, row[c4 * 32 + lane]);
    #pragma unroll
    for (int off = 16; off > 0; off >>= 1) mx = fmaxf(mx, __shfl_xor(mx, off, 32));
    float sum = 0.f;
    float ev[16];
    #pragma unroll
    for (int c4 = 0; c4 < 16; ++c4) {
      ev[c4] = __expf(row[c4 * 32 + lane] - mx);
      sum += ev[c4];
    }
    #pragma unroll
    for (int off = 16; off > 0; off >>= 1) sum += __shfl_xor(sum, off, 32);
    float rinv = 1.0f / sum;
    #pragma unroll
    for (int c4 = 0; c4 < 16; ++c4) prow[c4 * 32 + lane] = (__bf16)(ev[c4] * rinv);
  }
  __syncthreads();

  // O = P @ V : P (16x512 bf16 from LDS), V^T rows contiguous; 4 N-tiles of 16
  v8f o[4] = {(v8f){}, (v8f){}, (v8f){}, (v8f){}};
  const __bf16* prow = P + mr * NPP;
  for (int kk = 0; kk < NPP; kk += 32) {
    v16bf pa = frag16(prow + kk + half * 8, prow + kk + 16 + half * 8);
    #pragma unroll
    for (int ni = 0; ni < 4; ++ni) {
      const __bf16* vr = vbase + (size_t)(ni * 16 + mr) * NPP + kk + half * 8;
      o[ni] = wmma_bf16(pa, frag16(vr, vr + 16), o[ni]);
    }
  }

  const int pbase = p0 + 8 * half;
  __bf16* aop = ao + ((size_t)(b * NPP + pbase)) * DIMV + h * HD + mr;
  #pragma unroll
  for (int r = 0; r < 8; ++r)
    #pragma unroll
    for (int ni = 0; ni < 4; ++ni)
      aop[r * DIMV + ni * 16] = (__bf16)o[ni][r];
}

// ---------------- output projection + bias (fp32 out) ----------------
// grid (256, 2), 256 threads = 8 waves; wave -> 32x64 tile (8 WMMAs per K-step).
__global__ void __launch_bounds__(256)
k_proj(const __bf16* __restrict__ ao, const __bf16* __restrict__ pw,
       const float* __restrict__ pb, float* __restrict__ out) {
  const int lane = threadIdx.x & 31;
  const int wave = threadIdx.x >> 5;
  const int mr = lane & 15, half = lane >> 4;
  const int m_base = blockIdx.x * 32;
  const int j0     = (blockIdx.y * 8 + wave) * 64;

  const __bf16* ar[2];
  ar[0] = ao + (size_t)(m_base + mr) * DIMV;
  ar[1] = ar[0] + 16 * DIMV;
  const __bf16* br[4];
  #pragma unroll
  for (int ni = 0; ni < 4; ++ni) br[ni] = pw + (size_t)(j0 + ni * 16 + mr) * DIMV;

  v8f acc[2][4];
  #pragma unroll
  for (int mi = 0; mi < 2; ++mi)
    #pragma unroll
    for (int ni = 0; ni < 4; ++ni) acc[mi][ni] = (v8f){};

  for (int kk = 0; kk < DIMV; kk += 32) {
    v16bf a[2], f[4];
    #pragma unroll
    for (int mi = 0; mi < 2; ++mi)
      a[mi] = frag16(ar[mi] + kk + half * 8, ar[mi] + kk + 16 + half * 8);
    #pragma unroll
    for (int ni = 0; ni < 4; ++ni)
      f[ni] = frag16(br[ni] + kk + half * 8, br[ni] + kk + 16 + half * 8);
    #pragma unroll
    for (int mi = 0; mi < 2; ++mi)
      #pragma unroll
      for (int ni = 0; ni < 4; ++ni)
        acc[mi][ni] = wmma_bf16(a[mi], f[ni], acc[mi][ni]);
  }

  #pragma unroll
  for (int mi = 0; mi < 2; ++mi) {
    float* op = out + (size_t)(m_base + mi * 16 + 8 * half) * DIMV + j0 + mr;
    #pragma unroll
    for (int ni = 0; ni < 4; ++ni) {
      const float bias = pb[j0 + ni * 16 + mr];
      #pragma unroll
      for (int r = 0; r < 8; ++r) op[r * DIMV + ni * 16] = acc[mi][ni][r] + bias;
    }
  }
}

// ---------------- launch ----------------
extern "C" void kernel_launch(void* const* d_in, const int* in_sizes, int n_in,
                              void* d_out, int out_size, void* d_ws, size_t ws_size,
                              hipStream_t stream) {
  const float* x     = (const float*)d_in[0];
  const float* wq    = (const float*)d_in[1];
  const float* wk    = (const float*)d_in[2];
  const float* wv    = (const float*)d_in[3];
  const float* rpe_w = (const float*)d_in[4];
  const float* pw    = (const float*)d_in[5];
  const float* pb    = (const float*)d_in[6];
  float* out = (float*)d_out;

  char* ws = (char*)d_ws;
  const size_t SZ_X  = (size_t)BB * NPP * DIMV * 2;   // 16 MB each
  const size_t SZ_W  = (size_t)DIMV * DIMV * 2;       // 2 MB each
  __bf16* xb  = (__bf16*)(ws);
  __bf16* wqb = (__bf16*)(ws + SZ_X);
  __bf16* wkb = (__bf16*)(ws + SZ_X + SZ_W);
  __bf16* wvb = (__bf16*)(ws + SZ_X + 2 * SZ_W);
  __bf16* pwb = (__bf16*)(ws + SZ_X + 3 * SZ_W);
  __bf16* qhb = (__bf16*)(ws + SZ_X + 4 * SZ_W);
  __bf16* khb = (__bf16*)(ws + 2 * SZ_X + 4 * SZ_W);
  __bf16* vTb = (__bf16*)(ws + 3 * SZ_X + 4 * SZ_W);
  __bf16* aob = (__bf16*)(ws + 4 * SZ_X + 4 * SZ_W);
  unsigned char* bkt = (unsigned char*)(ws + 5 * SZ_X + 4 * SZ_W);

  // prep
  k_cvt_x<<<(BB * NPP * DIMV) / 256, 256, 0, stream>>>(xb, x);
  k_cvt_bf16<<<(DIMV * DIMV) / 256, 256, 0, stream>>>(wqb, wq, DIMV * DIMV);
  k_cvt_bf16<<<(DIMV * DIMV) / 256, 256, 0, stream>>>(wkb, wk, DIMV * DIMV);
  k_cvt_bf16<<<(DIMV * DIMV) / 256, 256, 0, stream>>>(wvb, wv, DIMV * DIMV);
  k_cvt_bf16<<<(DIMV * DIMV) / 256, 256, 0, stream>>>(pwb, pw, DIMV * DIMV);
  k_bucket<<<(NPP * NPP) / 256, 256, 0, stream>>>(bkt);

  // QKV projections: M=8192 in 32-row tiles, N=1024 in 32-col tiles, 8 waves/block
  k_qkv<<<dim3((BB * NPP) / 32, DIMV / (32 * 8)), 256, 0, stream>>>(
      xb, wqb, wkb, wvb, qhb, khb, vTb);

  // fused attention; LDS = 128KB scores + 64KB probs + 2KB lt = 194KB
  const size_t lds = (size_t)(4 * 16 * NPP) * 4 + (size_t)(4 * 16 * NPP) * 2 +
                     (size_t)(4 * 16 * 8) * 4;
  k_attn<<<BB * HH * (NPP / 64), 128, lds, stream>>>(qhb, khb, vTb, rpe_w, bkt, aob);

  // output projection: 32x64 per wave
  k_proj<<<dim3((BB * NPP) / 32, DIMV / (64 * 8)), 256, 0, stream>>>(aob, pwb, pb, out);
}
